// DeepHierarchicalNetwork_30803505447112
// MI455X (gfx1250) — matrix operand order repaired and checked by reading, not compiled, not run on hardware
//
#include <hip/hip_runtime.h>
#include <hip/hip_bf16.h>

// ---------------------------------------------------------------------------
// DeepHierarchicalNetwork for MI455X (gfx1250, wave32, WMMA).
// Compute-bound (~290 GFLOP over ~30MB L2-resident data) -> bf16 WMMA
// (v_wmma_f32_16x16x32_bf16, 8x MAC density of f32 WMMA) with fp32
// accumulation and fp32 gate math. GEMM tiles staged to LDS with the CDNA5
// async global->LDS path (double-buffered, ASYNCcnt) when the toolchain
// exposes it; sync ds_store staging otherwise.
// ---------------------------------------------------------------------------

typedef __attribute__((ext_vector_type(16))) __bf16 v16bf;
typedef __attribute__((ext_vector_type(8)))  float  v8f;

#define HID   512
#define H3    1536
#define BATCH 32
#define SEQ   128
#define ROWS  (BATCH * SEQ)   // 4096
#define DEPTHS 5
#define ARITY  4
#define OUTS   256
#define EPSG   1e-10f

#if defined(__AMDGCN__) && __has_builtin(__builtin_amdgcn_global_load_async_to_lds_b128) && \
    __has_builtin(__builtin_amdgcn_s_wait_asynccnt)
#define HAVE_ASYNC_LDS 1
#else
#define HAVE_ASYNC_LDS 0
#endif

#if HAVE_ASYNC_LDS
// Builtin signature (from hipcc diagnostic): param0 = AS(1) ptr to int
// ext_vector(4) [global src], param1 = AS(3) ptr to same [LDS dst], then two
// integer immediates (offset, cpol).
typedef int vi4_t __attribute__((ext_vector_type(4)));
typedef __attribute__((address_space(1))) vi4_t* gas_vi4p;
typedef __attribute__((address_space(3))) vi4_t* las_vi4p;
#define ASYNC_CP_B128(dst_lds, src_glb)                                          \
  __builtin_amdgcn_global_load_async_to_lds_b128((gas_vi4p)(src_glb),            \
                                                 (las_vi4p)(dst_lds), 0, 0)
#define ASYNC_WAIT0() __builtin_amdgcn_s_wait_asynccnt(0)
#endif

// ---------------- WMMA fragment helpers (layouts per ISA 7.12.2) -----------

// A fragment: 16x32 bf16 tile, row-major with leading dim ldk (halves).
// lane<16: M=lane, K = {0..7, 16..23}; lane>=16: M=lane-16, K = {8..15, 24..31}
__device__ __forceinline__ v16bf frag_a_bf(const __bf16* t, int ldk) {
  const int lane = threadIdx.x & 31;
  const __bf16* p = t + (lane & 15) * ldk + ((lane < 16) ? 0 : 8);
  v16bf f;
#pragma unroll
  for (int i = 0; i < 8; ++i) f[i] = p[i];
#pragma unroll
  for (int i = 0; i < 8; ++i) f[8 + i] = p[16 + i];
  return f;
}

// A fragment built from an fp32 LDS tile (converted on the fly).
__device__ __forceinline__ v16bf frag_a_f32(const float* h, int row0, int kb) {
  const int lane = threadIdx.x & 31;
  const float* p = h + (size_t)(row0 + (lane & 15)) * HID + kb + ((lane < 16) ? 0 : 8);
  v16bf f;
#pragma unroll
  for (int i = 0; i < 8; ++i) f[i] = (__bf16)p[i];
#pragma unroll
  for (int i = 0; i < 8; ++i) f[8 + i] = (__bf16)p[16 + i];
  return f;
}

// B fragment: B = W^T (32x16, KxN) from 16 rows of W (row-major, ld ldk).
// lane<16: N=lane, K = {0..7, 16..23}; lane>=16: N=lane-16, K = {8..15, 24..31}
__device__ __forceinline__ v16bf frag_b_rows(const __bf16* w, int ldk) {
  const int lane = threadIdx.x & 31;
  const __bf16* p = w + (size_t)(lane & 15) * ldk + ((lane < 16) ? 0 : 8);
  v16bf f;
#pragma unroll
  for (int i = 0; i < 8; ++i) f[i] = p[i];
#pragma unroll
  for (int i = 0; i < 8; ++i) f[8 + i] = p[16 + i];
  return f;
}

__device__ __forceinline__ v8f wmma_bf16(v16bf a, v16bf b, v8f c) {
  return __builtin_amdgcn_wmma_f32_16x16x32_bf16(false, a, false, b, (short)0, c,
                                                 false, false);
}

__device__ __forceinline__ float sigmoidf_(float x) { return 1.f / (1.f + expf(-x)); }

// ---------------- small utility kernels ------------------------------------

__global__ void k_convert(const float* __restrict__ s, __bf16* __restrict__ d, int n) {
  int i = blockIdx.x * blockDim.x + threadIdx.x;
  if (i < n) d[i] = (__bf16)s[i];
}

__global__ void k_init(const float* __restrict__ x, float* __restrict__ task,
                       __bf16* __restrict__ tb, float* __restrict__ nd) {
  size_t i = (size_t)blockIdx.x * blockDim.x + threadIdx.x;
  if (i < (size_t)ROWS * HID) {
    float v = x[i];
    task[i] = v;
    tb[i] = (__bf16)v;
  }
  if (blockIdx.x == 0 && threadIdx.x < BATCH) nd[threadIdx.x] = 1.f;
}

__global__ void k_zero_hs(float* __restrict__ hs, __bf16* __restrict__ hsb) {
  size_t i = (size_t)blockIdx.x * blockDim.x + threadIdx.x;
  if (i < (size_t)ROWS * HID) { hs[i] = 0.f; hsb[i] = (__bf16)0.f; }
}

// ---------------- gi = task_bf @ Wih^T + bih  (4096x1536, K=512) -----------
// 128x128 block tile, 8 waves (2x4), 64x32 wave tile, K staged 32-wide in a
// double-buffered LDS tile filled by async global->LDS copies.
__global__ __launch_bounds__(256) void k_gemm_gi(const __bf16* __restrict__ A,
                                                 const __bf16* __restrict__ W,
                                                 const float* __restrict__ bias,
                                                 float* __restrict__ C) {
  __shared__ uint4 sA[2][128][4];  // 2 x (128 rows x 32 halves)
  __shared__ uint4 sW[2][128][4];
  const int tid = threadIdx.x;
  const int lane = tid & 31, w = tid >> 5;
  const int wm = w >> 2, wn = w & 3;
  const int m0 = blockIdx.x * 128, n0 = blockIdx.y * 128;
  v8f acc[4][2];
#pragma unroll
  for (int im = 0; im < 4; ++im)
#pragma unroll
    for (int in = 0; in < 2; ++in) acc[im][in] = v8f{};

  const uint4* gA = (const uint4*)A;  // 8 halves per uint4, row stride HID/8
  const uint4* gW = (const uint4*)W;

  auto stage = [&](int buf, int kb) {
#pragma unroll
    for (int i = tid; i < 512; i += 256) {
      int r = i >> 2, c = i & 3;
      const uint4* srcA = gA + (size_t)(m0 + r) * (HID / 8) + (kb >> 3) + c;
      const uint4* srcW = gW + (size_t)(n0 + r) * (HID / 8) + (kb >> 3) + c;
#if HAVE_ASYNC_LDS
      ASYNC_CP_B128(&sA[buf][r][c], srcA);
      ASYNC_CP_B128(&sW[buf][r][c], srcW);
#else
      sA[buf][r][c] = *srcA;
      sW[buf][r][c] = *srcW;
#endif
    }
  };

  stage(0, 0);
  for (int it = 0; it < HID / 32; ++it) {
#if HAVE_ASYNC_LDS
    ASYNC_WAIT0();              // this wave's copies for buf[it&1] complete
#endif
    __syncthreads();            // everyone's copies visible / prior reads done
    if (it + 1 < HID / 32) stage((it + 1) & 1, (it + 1) * 32);
#if !HAVE_ASYNC_LDS
    __syncthreads();
#endif
    const __bf16* lA = (const __bf16*)sA[it & 1];
    const __bf16* lW = (const __bf16*)sW[it & 1];
    v16bf a[4], b[2];
#pragma unroll
    for (int im = 0; im < 4; ++im) a[im] = frag_a_bf(lA + (wm * 64 + im * 16) * 32, 32);
#pragma unroll
    for (int in = 0; in < 2; ++in) b[in] = frag_b_rows(lW + (wn * 32 + in * 16) * 32, 32);
#pragma unroll
    for (int im = 0; im < 4; ++im)
#pragma unroll
      for (int in = 0; in < 2; ++in) acc[im][in] = wmma_bf16(a[im], b[in], acc[im][in]);
#if !HAVE_ASYNC_LDS
    __syncthreads();
#endif
  }
#pragma unroll
  for (int in = 0; in < 2; ++in) {
    const int col = n0 + wn * 32 + in * 16 + (lane & 15);
    const float bv = bias[col];
#pragma unroll
    for (int im = 0; im < 4; ++im)
#pragma unroll
      for (int i = 0; i < 8; ++i) {
        int row = m0 + wm * 64 + im * 16 + ((lane < 16) ? i : 8 + i);
        C[(size_t)row * H3 + col] = acc[im][in][i] + bv;
      }
  }
}

// ---------------- full GRU scan in one kernel (grid=2: fwd, bwd) -----------
// h(32x512) fp32 lives in LDS; per step 16 waves compute gh = h @ Whh^T
// (32x1536, K=512) with bf16 WMMA, apply gates in fp32, update h in LDS.
__global__ __launch_bounds__(512) void k_gru_rec(
    const float* __restrict__ gi_f, const __bf16* __restrict__ Whh_f,
    const float* __restrict__ bhh_f, float* __restrict__ out_f,
    const float* __restrict__ gi_r, const __bf16* __restrict__ Whh_r,
    const float* __restrict__ bhh_r, float* __restrict__ out_r) {
  const bool rev = (blockIdx.x == 1);
  const float*  gi  = rev ? gi_r  : gi_f;
  const __bf16* Whh = rev ? Whh_r : Whh_f;
  const float*  bhh = rev ? bhh_r : bhh_f;
  float*        out = rev ? out_r : out_f;

  __shared__ float hl[BATCH][HID];  // 64 KB
  const int tid = threadIdx.x, lane = tid & 31, w = tid >> 5;  // 16 waves

  for (int i = tid; i < BATCH * HID; i += 512) ((float*)hl)[i] = 0.f;
  __syncthreads();

  for (int s = 0; s < SEQ; ++s) {
    const int sc = rev ? (SEQ - 1 - s) : s;
    float hnew[2][2][8];
#pragma unroll
    for (int jj = 0; jj < 2; ++jj) {
      const int j0 = (w * 2 + jj) * 16;  // 32 j-tiles over H, 2 per wave
      v8f acc[2][3];
#pragma unroll
      for (int mt = 0; mt < 2; ++mt)
#pragma unroll
        for (int g = 0; g < 3; ++g) acc[mt][g] = v8f{};
      for (int kb = 0; kb < HID; kb += 32) {
        v16bf a0 = frag_a_f32(&hl[0][0], 0, kb);
        v16bf a1 = frag_a_f32(&hl[0][0], 16, kb);
        v16bf br = frag_b_rows(Whh + (size_t)(j0)           * HID + kb, HID);
        v16bf bz = frag_b_rows(Whh + (size_t)(j0 + HID)     * HID + kb, HID);
        v16bf bn = frag_b_rows(Whh + (size_t)(j0 + 2 * HID) * HID + kb, HID);
        acc[0][0] = wmma_bf16(a0, br, acc[0][0]);
        acc[0][1] = wmma_bf16(a0, bz, acc[0][1]);
        acc[0][2] = wmma_bf16(a0, bn, acc[0][2]);
        acc[1][0] = wmma_bf16(a1, br, acc[1][0]);
        acc[1][1] = wmma_bf16(a1, bz, acc[1][1]);
        acc[1][2] = wmma_bf16(a1, bn, acc[1][2]);
      }
#pragma unroll
      for (int mt = 0; mt < 2; ++mt)
#pragma unroll
        for (int i = 0; i < 8; ++i) {
          int b = mt * 16 + ((lane < 16) ? i : 8 + i);
          int j = j0 + (lane & 15);
          const float* gir = gi + (size_t)(b * SEQ + sc) * H3;
          float r = sigmoidf_(gir[j]           + acc[mt][0][i] + bhh[j]);
          float z = sigmoidf_(gir[HID + j]     + acc[mt][1][i] + bhh[HID + j]);
          float n = tanhf(gir[2 * HID + j] + r * (acc[mt][2][i] + bhh[2 * HID + j]));
          hnew[jj][mt][i] = (1.f - z) * n + z * hl[b][j];
        }
    }
    __syncthreads();  // all reads of h done before overwrite
#pragma unroll
    for (int jj = 0; jj < 2; ++jj) {
      int j = (w * 2 + jj) * 16 + (lane & 15);
#pragma unroll
      for (int mt = 0; mt < 2; ++mt)
#pragma unroll
        for (int i = 0; i < 8; ++i) {
          int b = mt * 16 + ((lane < 16) ? i : 8 + i);
          hl[b][j] = hnew[jj][mt][i];
        }
    }
    __syncthreads();
  }
  for (int i = tid; i < BATCH * HID; i += 512) out[i] = ((float*)hl)[i];
}

// ---------------- logits + gumbel straight-through gate --------------------
__global__ void k_gate_select(const float* __restrict__ hf, const float* __restrict__ hb,
                              const float* __restrict__ lw, const float* __restrict__ lb,
                              const float* __restrict__ gu, float* __restrict__ nd,
                              float* __restrict__ done) {
  __shared__ float lg[BATCH][2];
  int t = threadIdx.x;
  if (t < 64) {
    int b = t >> 1, c = t & 1;
    float s = lb[c];
    for (int h = 0; h < HID; ++h) s += (hf[b * HID + h] + hb[b * HID + h]) * lw[c * HID + h];
    lg[b][c] = s;
  }
  __syncthreads();
  if (t < BATCH) {
    float g0 = -logf(-logf(gu[t * 2 + 0] + EPSG) + EPSG);
    float g1 = -logf(-logf(gu[t * 2 + 1] + EPSG) + EPSG);
    float s0 = lg[t][0] + g0, s1 = lg[t][1] + g1;
    float hard1 = (s1 > s0) ? 1.f : 0.f;  // argmax ties -> class 0, like jnp.argmax
    float ndn = nd[t] * hard1;            // straight-through value == one-hot
    nd[t] = ndn;
    done[t] = 1.f - ndn;
  }
}

// ---------------- fused splitter-GRU step ----------------------------------
// Block owns (128 rows) x (16 j-cols); computes r/z/n accumulators (cols j,
// j+512, j+1024) together so gh is never materialized. bf16 h is
// double-buffered across steps (cross-block RAW safety). LDS tiles are
// double-buffered and filled with async copies.
__global__ __launch_bounds__(256) void k_split_step(
    const __bf16* __restrict__ hin_bf, __bf16* __restrict__ hout_bf,
    float* __restrict__ hs, const float* __restrict__ gi,
    const __bf16* __restrict__ Whh, const float* __restrict__ bhh) {
  __shared__ uint4 sA[2][128][4];  // h rows
  __shared__ uint4 sW[2][48][4];   // 3 gates x 16 W-rows x 32 halves
  const int tid = threadIdx.x, lane = tid & 31, w = tid >> 5;  // 8 waves
  const int m0 = blockIdx.x * 128;
  const int j0 = blockIdx.y * 16;
  v8f acc[3] = {v8f{}, v8f{}, v8f{}};

  const uint4* gA = (const uint4*)hin_bf;
  const uint4* gW = (const uint4*)Whh;

  auto stage = [&](int buf, int kb) {
#pragma unroll
    for (int i = tid; i < 512; i += 256) {
      int r = i >> 2, c = i & 3;
      const uint4* src = gA + (size_t)(m0 + r) * (HID / 8) + (kb >> 3) + c;
#if HAVE_ASYNC_LDS
      ASYNC_CP_B128(&sA[buf][r][c], src);
#else
      sA[buf][r][c] = *src;
#endif
    }
    if (tid < 192) {
      int row = tid >> 2, c = tid & 3;
      int g = row >> 4, r = row & 15;
      const uint4* src = gW + (size_t)(j0 + g * HID + r) * (HID / 8) + (kb >> 3) + c;
#if HAVE_ASYNC_LDS
      ASYNC_CP_B128(&sW[buf][row][c], src);
#else
      sW[buf][row][c] = *src;
#endif
    }
  };

  stage(0, 0);
  for (int it = 0; it < HID / 32; ++it) {
#if HAVE_ASYNC_LDS
    ASYNC_WAIT0();
#endif
    __syncthreads();
    if (it + 1 < HID / 32) stage((it + 1) & 1, (it + 1) * 32);
#if !HAVE_ASYNC_LDS
    __syncthreads();
#endif
    v16bf a = frag_a_bf((const __bf16*)sA[it & 1] + (w * 16) * 32, 32);
#pragma unroll
    for (int g = 0; g < 3; ++g) {
      v16bf b = frag_b_rows((const __bf16*)sW[it & 1] + (g * 16) * 32, 32);
      acc[g] = wmma_bf16(a, b, acc[g]);
    }
#if !HAVE_ASYNC_LDS
    __syncthreads();
#endif
  }
#pragma unroll
  for (int i = 0; i < 8; ++i) {
    int row = m0 + w * 16 + ((lane < 16) ? i : 8 + i);
    int j = j0 + (lane & 15);
    const float* gir = gi + (size_t)row * H3;
    float r = sigmoidf_(gir[j]       + acc[0][i] + bhh[j]);
    float z = sigmoidf_(gir[HID + j] + acc[1][i] + bhh[HID + j]);
    float n = tanhf(gir[2 * HID + j] + r * (acc[2][i] + bhh[2 * HID + j]));
    float hold = hs[(size_t)row * HID + j];
    float hn = (1.f - z) * n + z * hold;
    hs[(size_t)row * HID + j] = hn;
    hout_bf[(size_t)row * HID + j] = (__bf16)hn;
  }
}

// ---------------- task = done*task + nd*sub --------------------------------
__global__ void k_task_update(float* __restrict__ task, __bf16* __restrict__ tb,
                              const float* __restrict__ hs, const float* __restrict__ nd,
                              const float* __restrict__ done) {
  size_t i = (size_t)blockIdx.x * blockDim.x + threadIdx.x;
  if (i >= (size_t)ROWS * HID) return;
  int row = (int)(i >> 9);   // /HID
  int b = row >> 7;          // /SEQ
  float v = done[b] * task[i] + nd[b] * hs[i];
  task[i] = v;
  tb[i] = (__bf16)v;
}

// ---------------- final reduction + projection -----------------------------
__global__ __launch_bounds__(256) void k_colsum(const float* __restrict__ task,
                                                float* __restrict__ q) {
  __shared__ float red[256];
  int h = blockIdx.x;
  float s = 0.f;
  for (int r = threadIdx.x; r < ROWS; r += 256) s += task[(size_t)r * HID + h];
  red[threadIdx.x] = s;
  __syncthreads();
  for (int st = 128; st > 0; st >>= 1) {
    if (threadIdx.x < st) red[threadIdx.x] += red[threadIdx.x + st];
    __syncthreads();
  }
  if (threadIdx.x == 0) q[h] = red[0] * (1.f / SEQ);
}

__global__ void k_out(const float* __restrict__ q, const float* __restrict__ oW,
                      const float* __restrict__ ob, float* __restrict__ out) {
  int o = threadIdx.x;
  if (o >= OUTS) return;
  float s = (float)BATCH * ob[o];
  for (int h = 0; h < HID; ++h) s += q[h] * oW[o * HID + h];
  out[o] = s;
}

// ---------------------------------------------------------------------------
extern "C" void kernel_launch(void* const* d_in, const int* in_sizes, int n_in,
                              void* d_out, int out_size, void* d_ws, size_t ws_size,
                              hipStream_t stream) {
  (void)in_sizes; (void)n_in; (void)out_size; (void)ws_size;
  const float* x        = (const float*)d_in[0];
  const float* gu       = (const float*)d_in[1];
  const float* ts_Wih   = (const float*)d_in[2];
  const float* ts_Whh   = (const float*)d_in[3];
  const float* ts_bih   = (const float*)d_in[4];
  const float* ts_bhh   = (const float*)d_in[5];
  const float* tgf_Wih  = (const float*)d_in[6];
  const float* tgf_Whh  = (const float*)d_in[7];
  const float* tgf_bih  = (const float*)d_in[8];
  const float* tgf_bhh  = (const float*)d_in[9];
  const float* tgb_Wih  = (const float*)d_in[10];
  const float* tgb_Whh  = (const float*)d_in[11];
  const float* tgb_bih  = (const float*)d_in[12];
  const float* tgb_bhh  = (const float*)d_in[13];
  const float* lw       = (const float*)d_in[14];
  const float* lb       = (const float*)d_in[15];
  const float* oW       = (const float*)d_in[16];
  const float* ob       = (const float*)d_in[17];
  float* out = (float*)d_out;

  char* ws = (char*)d_ws;
  size_t off = 0;
  auto alloc = [&](size_t bytes) -> void* {
    void* p = ws + off;
    off = (off + bytes + 255) & ~(size_t)255;
    return p;
  };
  float*  task    = (float*) alloc((size_t)ROWS * HID * 4);
  __bf16* task_bf = (__bf16*)alloc((size_t)ROWS * HID * 2);
  float*  gi_f    = (float*) alloc((size_t)ROWS * H3 * 4);
  float*  gi_b    = (float*) alloc((size_t)ROWS * H3 * 4);
  float*  gi_s    = gi_f;  // reused after the recurrences consume gi_f
  float*  hs      = (float*) alloc((size_t)ROWS * HID * 4);
  __bf16* hsb0    = (__bf16*)alloc((size_t)ROWS * HID * 2);
  __bf16* hsb1    = (__bf16*)alloc((size_t)ROWS * HID * 2);
  const size_t WN = (size_t)H3 * HID;
  __bf16* wb_ts_ih  = (__bf16*)alloc(WN * 2);
  __bf16* wb_ts_hh  = (__bf16*)alloc(WN * 2);
  __bf16* wb_tgf_ih = (__bf16*)alloc(WN * 2);
  __bf16* wb_tgf_hh = (__bf16*)alloc(WN * 2);
  __bf16* wb_tgb_ih = (__bf16*)alloc(WN * 2);
  __bf16* wb_tgb_hh = (__bf16*)alloc(WN * 2);
  float*  hf   = (float*)alloc(BATCH * HID * 4);
  float*  hb   = (float*)alloc(BATCH * HID * 4);
  float*  nd   = (float*)alloc(BATCH * 4);
  float*  done = (float*)alloc(BATCH * 4);
  float*  q    = (float*)alloc(HID * 4);

  const int cb = (int)((WN + 255) / 256);
  k_convert<<<cb, 256, 0, stream>>>(ts_Wih,  wb_ts_ih,  (int)WN);
  k_convert<<<cb, 256, 0, stream>>>(ts_Whh,  wb_ts_hh,  (int)WN);
  k_convert<<<cb, 256, 0, stream>>>(tgf_Wih, wb_tgf_ih, (int)WN);
  k_convert<<<cb, 256, 0, stream>>>(tgf_Whh, wb_tgf_hh, (int)WN);
  k_convert<<<cb, 256, 0, stream>>>(tgb_Wih, wb_tgb_ih, (int)WN);
  k_convert<<<cb, 256, 0, stream>>>(tgb_Whh, wb_tgb_hh, (int)WN);

  const int eb = (ROWS * HID + 255) / 256;
  k_init<<<eb, 256, 0, stream>>>(x, task, task_bf, nd);

  const dim3 gg(ROWS / 128, H3 / 128);  // 32 x 12
  for (int d = 0; d < DEPTHS; ++d) {
    k_gemm_gi<<<gg, 256, 0, stream>>>(task_bf, wb_tgf_ih, tgf_bih, gi_f);
    k_gemm_gi<<<gg, 256, 0, stream>>>(task_bf, wb_tgb_ih, tgb_bih, gi_b);
    k_gru_rec<<<2, 512, 0, stream>>>(gi_f, wb_tgf_hh, tgf_bhh, hf,
                                     gi_b, wb_tgb_hh, tgb_bhh, hb);
    k_gate_select<<<1, 64, 0, stream>>>(hf, hb, lw, lb, gu + (size_t)d * BATCH * 2,
                                        nd, done);
    k_gemm_gi<<<gg, 256, 0, stream>>>(task_bf, wb_ts_ih, ts_bih, gi_s);
    k_zero_hs<<<eb, 256, 0, stream>>>(hs, hsb0);
    __bf16* bufs[2] = {hsb0, hsb1};
    for (int t = 0; t < ARITY; ++t)
      k_split_step<<<dim3(ROWS / 128, HID / 16), 256, 0, stream>>>(
          bufs[t & 1], bufs[(t + 1) & 1], hs, gi_s, wb_ts_hh, ts_bhh);
    k_task_update<<<eb, 256, 0, stream>>>(task, task_bf, hs, nd, done);
  }
  k_colsum<<<HID, 256, 0, stream>>>(task, q);
  k_out<<<1, 256, 0, stream>>>(q, oW, ob, out);
}
